// Phenotype_5686536700208
// MI455X (gfx1250) — matrix operand order repaired and checked by reading, not compiled, not run on hardware
//
#include <hip/hip_runtime.h>
#include <stdint.h>
#include <stddef.h>

// ---- genotype constants (match reference) ----
#define N_NODES 100000
#define N_IN    1024
#define N_OUT   1024
#define FAN_IN  32
#define BATCH   128
#define CHUNK   16
#define NCHUNK  ((N_NODES - N_IN) / CHUNK)   // 6186 (exact)
#define MAIN_WGS 64

typedef __attribute__((ext_vector_type(16))) __bf16         v16bf;
typedef __attribute__((ext_vector_type(8)))  float          v8f;
typedef __attribute__((ext_vector_type(8)))  unsigned       v8u;

__device__ __forceinline__ unsigned short f32_to_bf16_raw(float f) {
  union { float f; unsigned u; } x; x.f = f;
  unsigned u = x.u;
  u += 0x7FFFu + ((u >> 16) & 1u);     // round-to-nearest-even
  return (unsigned short)(u >> 16);
}
__device__ __forceinline__ float bf16_raw_to_f32(unsigned short h) {
  union { unsigned u; float f; } x; x.u = ((unsigned)h) << 16;
  return x.f;
}

// ---- kernel 1: init vals[0:1024] = x^T (as bf16), zero chunk metadata, progress=N_IN ----
__global__ void phen_init(const float* __restrict__ x, unsigned short* __restrict__ vals,
                          int* __restrict__ maxdep, unsigned* __restrict__ flags,
                          int* __restrict__ progress) {
  int idx = blockIdx.x * blockDim.x + threadIdx.x;
  if (idx < N_IN * BATCH) {
    int b = idx / N_IN;
    int node = idx - b * N_IN;
    vals[(unsigned)(node * BATCH + b)] = f32_to_bf16_raw(x[(unsigned)(b * N_IN + node)]);
  }
  if (idx < NCHUNK) { maxdep[idx] = 0; flags[idx] = 0u; }
  if (idx == 0) *progress = N_IN;
}

// ---- kernel 2: per-chunk dependency scan ----
// maxdep[c] = max predecessor id OUTSIDE the chunk (cross-chunk wait target)
// flags[c]  = bitmask of nodes with same-chunk predecessors (serial fixup)
__global__ void phen_deps(const int* __restrict__ I, int* __restrict__ maxdep,
                          unsigned* __restrict__ flags) {
  int gid = blockIdx.x * blockDim.x + threadIdx.x;
  if (gid >= N_NODES - N_IN) return;
  const int node  = N_IN + gid;
  const int chunk = gid >> 4;
  const int nb    = N_IN + chunk * CHUNK;
  const int4* __restrict__ Iv = (const int4*)(I + (size_t)node * FAN_IN);
  int mx = -1; unsigned fl = 0u;
#pragma unroll
  for (int q = 0; q < FAN_IN / 4; ++q) {
    int4 v = Iv[q];
    int ids4[4] = {v.x, v.y, v.z, v.w};
#pragma unroll
    for (int j = 0; j < 4; ++j) {
      int id = ids4[j];
      if (id >= nb) fl = 1u;
      else          mx = id > mx ? id : mx;
    }
  }
  if (mx >= 0) atomicMax(&maxdep[chunk], mx);
  if (fl)      atomicOr(&flags[chunk], 1u << (gid & 15));
}

// ---- kernel 3: persistent dataflow evaluator, WMMA bf16 16x16x32 (K == FAN_IN) ----
__global__ __launch_bounds__(256, 1)
void phen_main(const float* __restrict__ W, const int* __restrict__ I,
               unsigned short* __restrict__ vals, const int* __restrict__ maxdep,
               const unsigned* __restrict__ flags, int* __restrict__ progress) {
  __shared__ __align__(16) float sW[CHUNK * FAN_IN];   // chunk weights (f32)
  __shared__ __align__(16) int   sI[CHUNK * FAN_IN];   // chunk predecessor ids

  const int tid   = threadIdx.x;
  const int lane  = tid & 31;
  const int wave  = tid >> 5;
  const int khalf = lane >> 4;                 // 0: lanes 0-15, 1: lanes 16-31
  const int myrow = lane & 15;                 // A-matrix row owned by this lane
  const unsigned col = (unsigned)(wave * 16 + myrow);  // batch column owned by this lane

  const unsigned short* __restrict__ vbase = vals;     // uniform SGPR base for gathers

  for (int chunk = blockIdx.x; chunk < NCHUNK; chunk += MAIN_WGS) {
    const int nb = N_IN + chunk * CHUNK;

    // stage this chunk's weights + ids into LDS (2 KB + 2 KB)
    for (int t = tid; t < CHUNK * FAN_IN; t += 256) {
      sW[t] = W[(size_t)nb * FAN_IN + t];
      sI[t] = I[(size_t)nb * FAN_IN + t];
    }
    // prefetch our next chunk's weights/ids (global_prefetch_b8)
    if (tid == 0) {
      int nc = chunk + MAIN_WGS;
      if (nc < NCHUNK) {
        const size_t noff = (size_t)(N_IN + nc * CHUNK) * FAN_IN;
        __builtin_prefetch(&W[noff], 0, 1);
        __builtin_prefetch(&I[noff], 0, 1);
      }
    }
    // wait until every cross-chunk dependency is committed
    if (tid == 0) {
      const int need = maxdep[chunk] + 1;
      while (__hip_atomic_load(progress, __ATOMIC_ACQUIRE, __HIP_MEMORY_SCOPE_AGENT) < need)
        __builtin_amdgcn_s_sleep(1);
    }
    __syncthreads();

    // per-lane A weights (row = lane&15) built once per chunk, as raw dwords
    // A 16-bit layout (matches B): lanes0-15 -> K 0-15 ; lanes16-31 -> K 16-31
    v8u awu;
    {
      v16bf aw;
#pragma unroll
      for (int e = 0; e < 16; ++e) {
        const int ka = e + khalf * 16;
        aw[e] = __builtin_bit_cast(__bf16, f32_to_bf16_raw(sW[myrow * FAN_IN + ka]));
      }
      awu = __builtin_bit_cast(v8u, aw);
    }

    // 16 block-diagonal WMMAs accumulate the [16 node x 16 batch] f32 tile:
    // A_i = aw masked to row i only; B_i = node i's gathered 32x16 operand.
    // unroll 4: keep up to 4 nodes' gather clauses in flight to hide L2 latency
    // (the WMMA chain is serialized through the accumulator anyway).
    v8f acc = {0.f, 0.f, 0.f, 0.f, 0.f, 0.f, 0.f, 0.f};
#pragma unroll 4
    for (int i = 0; i < CHUNK; ++i) {
      // mask A to zero except on this lane's iteration (no EXEC divergence)
      const unsigned m = (myrow == i) ? 0xFFFFFFFFu : 0u;
      const v8u mv = {m, m, m, m, m, m, m, m};
      const v16bf a = __builtin_bit_cast(v16bf, awu & mv);

      // this lane's 16 predecessor ids as 4x ds_load_b128
      // B 32x16, k order chosen here (same as A): lanes0-15 -> K 0-15 ; lanes16-31 -> K 16-31 ; N = lane&15
      const int4* idv = (const int4*)&sI[i * FAN_IN + khalf * 16];
      const int4 q0 = idv[0], q1 = idv[1], q2 = idv[2], q3 = idv[3];
      const int ids16[16] = {q0.x, q0.y, q0.z, q0.w, q1.x, q1.y, q1.z, q1.w,
                             q2.x, q2.y, q2.z, q2.w, q3.x, q3.y, q3.z, q3.w};

      v16bf b;
#pragma unroll
      for (int e = 0; e < 16; ++e) {
        // 32-bit element offset (max 100000*128+127 < 2^31): SGPR base + u32 voffset
        const unsigned off = ((unsigned)ids16[e] << 7) + col;
        b[e] = __builtin_bit_cast(__bf16, vbase[off]);
      }

      acc = __builtin_amdgcn_wmma_f32_16x16x32_bf16(false, a, false, b,
                                                    (short)0, acc, false, false);
    }

    // ReLU + store tile. D layout: VGPR r -> row (r + 8*khalf), N = lane&15
    const unsigned mrow = (unsigned)(khalf * 8);
#pragma unroll
    for (int r = 0; r < 8; ++r) {
      float v = acc[r];
      v = v > 0.f ? v : 0.f;
      vals[(unsigned)(nb + mrow + r) * (unsigned)BATCH + col] = f32_to_bf16_raw(v);
    }
    __threadfence();
    __syncthreads();

    // serial fixup of nodes with intra-chunk dependencies (ascending order), wave 0 only
    if (wave == 0) {
      unsigned fl = flags[chunk];
      while (fl) {
        const int i = __builtin_ctz(fl);
        fl &= fl - 1u;
        for (int b2 = lane; b2 < BATCH; b2 += 32) {
          float s = 0.f;
#pragma unroll
          for (int k = 0; k < FAN_IN; ++k) {
            const unsigned off = ((unsigned)sI[i * FAN_IN + k] << 7) + (unsigned)b2;
            const unsigned short rv = ((volatile const unsigned short*)vals)[off];
            s += sW[i * FAN_IN + k] * bf16_raw_to_f32(rv);
          }
          s = s > 0.f ? s : 0.f;
          ((volatile unsigned short*)vals)[(unsigned)(nb + i) * (unsigned)BATCH + (unsigned)b2] =
              f32_to_bf16_raw(s);
        }
        // same-wave store->load to same address is ordered (ISA 7.3); volatile stops caching
      }
    }
    __threadfence();
    __syncthreads();

    // ordered commit: publish nodes [0, nb+16) as done
    if (tid == 0) {
      while (__hip_atomic_load(progress, __ATOMIC_ACQUIRE, __HIP_MEMORY_SCOPE_AGENT) != nb)
        __builtin_amdgcn_s_sleep(1);
      __hip_atomic_store(progress, nb + CHUNK, __ATOMIC_RELEASE, __HIP_MEMORY_SCOPE_AGENT);
    }
    __syncthreads();
  }
}

// ---- kernel 4: emit output [B, N_OUT] in f32 ----
__global__ void phen_out(const unsigned short* __restrict__ vals, float* __restrict__ out) {
  int idx = blockIdx.x * blockDim.x + threadIdx.x;
  if (idx >= N_OUT * BATCH) return;
  int b = idx / N_OUT;
  int j = idx - b * N_OUT;
  out[(unsigned)(b * N_OUT + j)] =
      bf16_raw_to_f32(vals[(unsigned)(N_NODES - N_OUT + j) * (unsigned)BATCH + (unsigned)b]);
}

extern "C" void kernel_launch(void* const* d_in, const int* in_sizes, int n_in,
                              void* d_out, int out_size, void* d_ws, size_t ws_size,
                              hipStream_t stream) {
  (void)in_sizes; (void)n_in; (void)out_size; (void)ws_size;
  const float* x = (const float*)d_in[0];   // [128, 1024]
  const float* W = (const float*)d_in[1];   // [100000, 32]
  const int*   I = (const int*)d_in[2];     // [100000, 32]
  float* out = (float*)d_out;               // [128, 1024]

  // workspace layout (~25.65 MB)
  char* ws = (char*)d_ws;
  unsigned short* vals = (unsigned short*)ws;                       // N_NODES*128 bf16
  size_t off = (size_t)N_NODES * BATCH * sizeof(unsigned short);
  int* maxdep = (int*)(ws + off);       off += sizeof(int) * NCHUNK;
  unsigned* flags = (unsigned*)(ws + off); off += sizeof(unsigned) * NCHUNK;
  int* progress = (int*)(ws + off);

  phen_init<<<(N_IN * BATCH + 255) / 256, 256, 0, stream>>>(x, vals, maxdep, flags, progress);
  phen_deps<<<((N_NODES - N_IN) + 255) / 256, 256, 0, stream>>>(I, maxdep, flags);
  phen_main<<<MAIN_WGS, 256, 0, stream>>>(W, I, vals, maxdep, flags, progress);
  phen_out<<<(N_OUT * BATCH + 255) / 256, 256, 0, stream>>>(vals, out);
}
